// SparseBEVSampling_12446815224514
// MI455X (gfx1250) — matrix operand order repaired and checked by reading, not compiled, not run on hardware
//
#include <hip/hip_runtime.h>
#include <hip/hip_bf16.h>
#include <math.h>

// ---- problem constants ----
#define QN    10000
#define GN    4
#define PN    8
#define ZN    4
#define LN    4
#define NV    6
#define DN    256
#define CN    64          // D / G
#define NOFF  96          // G*P*3
#define NSW   128         // G*P*L
#define IMGW  704.0f
#define IMGH  256.0f
#define EPSF  1e-5f
#define MTILES_PER_WAVE 5   // 625 M-tiles / 125 blocks

typedef __attribute__((ext_vector_type(16))) _Float16 v16h;
typedef __attribute__((ext_vector_type(8)))  float    v8f;

// --------------------------------------------------------------------------
// Kernel 1: GEMM  out = query @ W + b   via v_wmma_f32_16x16x32_f16.
// NW is a compile-time constant so all B-matrix strided loads become
// immediate-offset global_load_b32 in one clause (no address chains).
// Full-K B panel (8 chunks x v16h = 64 VGPRs) is cached in registers and
// reused across MTILES_PER_WAVE M-tiles.
// grid = (125, NW/16), block = 32 (one wave per tile column).
// --------------------------------------------------------------------------
template <int NW>
__global__ __launch_bounds__(32)
void sbev_gemm_wmma(const float* __restrict__ query,
                    const float* __restrict__ W,
                    const float* __restrict__ bias,
                    float* __restrict__ out)
{
    const int lane = threadIdx.x;        // 0..31
    const int half = lane >> 4;          // 0/1
    const int l16  = lane & 15;
    const int n0   = blockIdx.y * 16;
    const int ncol = n0 + l16;

    // ---- preload full-K B panel into registers (compile-time offsets) ----
    // chunk kc covers K = kc*32 .. kc*32+31; lanes 0-15: K-lo 16, 16-31: K-hi 16
    v16h bfrag[8];
    const float* wp = W + (half << 4) * NW + ncol;
    #pragma unroll
    for (int kc = 0; kc < 8; ++kc) {
        #pragma unroll
        for (int j = 0; j < 8; ++j) {
            bfrag[kc][2 * j]     = (_Float16)wp[(kc * 32 + 2 * j    ) * NW];
            bfrag[kc][2 * j + 1] = (_Float16)wp[(kc * 32 + 2 * j + 1) * NW];
        }
    }

    const float bv = bias[ncol];

    // ---- sweep M-tiles, reusing the B registers ----
    #pragma unroll 1
    for (int i = 0; i < MTILES_PER_WAVE; ++i) {
        const int mt   = blockIdx.x * MTILES_PER_WAVE + i;   // 0..624
        const int mrow = mt * 16 + l16;
        const float* qrow = query + (long long)mrow * DN;

        v8f acc = {};
        #pragma unroll
        for (int kc = 0; kc < 8; ++kc) {
            const int k0 = kc * 32;
            v16h a;
            // A: 16x32 f16. VGPR j 0..7: K = half*16 + 2j.
            #pragma unroll
            for (int j = 0; j < 8; ++j) {
                const int k = k0 + (half << 4) + (j << 1);
                const float2 q2 = *(const float2*)(qrow + k);
                a[2 * j]     = (_Float16)q2.x;
                a[2 * j + 1] = (_Float16)q2.y;
            }
            acc = __builtin_amdgcn_wmma_f32_16x16x32_f16(
                      false, a, false, bfrag[kc], (short)0, acc, false, false);
        }

        // C/D layout: lane col = l16; VGPR r holds row m = r + 8*half.
        #pragma unroll
        for (int r = 0; r < 8; ++r) {
            const int m = mt * 16 + r + (half << 3);
            out[m * NW + ncol] = acc[r] + bv;
        }
    }
}

// --------------------------------------------------------------------------
// Kernel 2: per-(q,g,p) projection + view select + softmax.  320k threads.
// --------------------------------------------------------------------------
__device__ __forceinline__ float sigm(float x) { return 1.0f / (1.0f + expf(-x)); }

__global__ __launch_bounds__(256)
void sbev_proj(const float* __restrict__ refp, const float* __restrict__ l2i,
               const float* __restrict__ off_raw, const float* __restrict__ sw_raw,
               float* __restrict__ uvn, float* __restrict__ wts,
               float* __restrict__ spc_out)
{
    const int tid = blockIdx.x * blockDim.x + threadIdx.x;   // (q*G + g)*P + p
    if (tid >= QN * GN * PN) return;
    const int q = tid >> 5;
    const int g = (tid >> 3) & 3;
    const int p = tid & 7;
    const int z = p >> 1;

    // offset transform: sigmoid(xy)*4 - 2, sigmoid(z)*2 - 1
    const float* o = off_raw + q * NOFF + (g * 8 + p) * 3;
    const float ox = 4.0f * sigm(o[0]) - 2.0f;
    const float oy = 4.0f * sigm(o[1]) - 2.0f;
    const float oz = 2.0f * sigm(o[2]) - 1.0f;

    const float* rp = refp + q * (ZN * 3) + z * 3;
    const float px = rp[0] * 102.4f - 51.2f + ox;
    const float py = rp[1] * 102.4f - 51.2f + oy;
    const float pz = rp[2] * 8.0f   - 5.0f  + oz;

    float bu = 0.0f, bv = 0.0f, u0 = 0.0f, v0 = 0.0f;
    int   bview = 0;
    bool  found = false;
    #pragma unroll
    for (int n = 0; n < NV; ++n) {
        const float* M = l2i + n * 16;
        const float c0 = M[0] * px + M[1] * py + M[2]  * pz + M[3];
        const float c1 = M[4] * px + M[5] * py + M[6]  * pz + M[7];
        const float c2 = M[8] * px + M[9] * py + M[10] * pz + M[11];
        const float h  = fmaxf(c2, EPSF);
        const float u  = (c0 / h) * (1.0f / IMGW);
        const float v  = (c1 / h) * (1.0f / IMGH);
        const bool valid = (c2 > EPSF) && (u > 0.0f) && (u < 1.0f)
                                       && (v > 0.0f) && (v < 1.0f);
        if (n == 0) { u0 = u; v0 = v; }
        if (valid && !found) { found = true; bview = n; bu = u; bv = v; }
    }
    if (!found) { bview = 0; bu = u0; bv = v0; }

    // spc[..., :2] output segment, layout (B*G, Q, P, 2)
    const int sidx = ((g * QN + q) * PN + p) * 2;
    spc_out[sidx]     = bu;
    spc_out[sidx + 1] = bv;

    uvn[tid * 3]     = bu;
    uvn[tid * 3 + 1] = bv;
    uvn[tid * 3 + 2] = (float)bview;

    // softmax over L=4 sampling weights
    const float* lg = sw_raw + q * NSW + (g * 8 + p) * 4;
    const float mx = fmaxf(fmaxf(lg[0], lg[1]), fmaxf(lg[2], lg[3]));
    const float e0 = expf(lg[0] - mx), e1 = expf(lg[1] - mx);
    const float e2 = expf(lg[2] - mx), e3 = expf(lg[3] - mx);
    const float inv = 1.0f / (e0 + e1 + e2 + e3);
    float4 w4 = { e0 * inv, e1 * inv, e2 * inv, e3 * inv };
    *(float4*)(wts + tid * 4) = w4;
}

// --------------------------------------------------------------------------
// Kernel 3: weighted multi-level bilinear gather.  One thread per (q,g,p,c).
// Channels are the fastest index -> coalesced output, broadcast parameters.
// --------------------------------------------------------------------------
__global__ __launch_bounds__(256)
void sbev_sample(const float* __restrict__ f0, const float* __restrict__ f1,
                 const float* __restrict__ f2, const float* __restrict__ f3,
                 const float* __restrict__ uvn, const float* __restrict__ wts,
                 float* __restrict__ out)
{
    const long long idx = (long long)blockIdx.x * blockDim.x + threadIdx.x;
    if (idx >= (long long)QN * GN * PN * CN) return;
    const int c   = (int)(idx & (CN - 1));
    const int tid = (int)(idx >> 6);          // (q*G+g)*P + p
    const int g   = (tid >> 3) & 3;

    const float2 uv = *(const float2*)(uvn + tid * 3);
    const float u    = uv.x;
    const float v    = uv.y;
    const int   view = (int)uvn[tid * 3 + 2];
    const float4 w4  = *(const float4*)(wts + tid * 4);
    const float swl_arr[LN] = { w4.x, w4.y, w4.z, w4.w };

    const float* feats[LN] = { f0, f1, f2, f3 };
    const int HH[LN] = { 32, 16, 8, 4 };
    const int WW[LN] = { 88, 44, 22, 11 };

    float acc = 0.0f;
    #pragma unroll
    for (int l = 0; l < LN; ++l) {
        const int H = HH[l], W = WW[l];
        const float pxl = u * (float)(W - 1);
        const float pyl = v * (float)(H - 1);
        const float x0f = floorf(pxl), y0f = floorf(pyl);
        const float wx = pxl - x0f,   wy = pyl - y0f;
        const int x0 = (int)x0f, y0 = (int)y0f;
        // feat layout: (BG, C, N, H, W)
        const float* base = feats[l]
            + ((long long)((g * CN + c) * NV + view)) * (long long)(H * W);

        float s = 0.0f;
        #pragma unroll
        for (int cy = 0; cy < 2; ++cy) {
            #pragma unroll
            for (int cx = 0; cx < 2; ++cx) {
                const int xi = x0 + cx, yi = y0 + cy;
                const bool m = (xi >= 0) && (xi < W) && (yi >= 0) && (yi < H);
                const int xc = xi < 0 ? 0 : (xi > W - 1 ? W - 1 : xi);
                const int yc = yi < 0 ? 0 : (yi > H - 1 ? H - 1 : yi);
                const float wgt = (cx ? wx : 1.0f - wx) * (cy ? wy : 1.0f - wy);
                s += base[yc * W + xc] * (m ? wgt : 0.0f);
            }
        }
        acc += swl_arr[l] * s;
    }
    out[idx] = acc;    // final (B,Q,G,P,C) flat == tid*64 + c
}

// --------------------------------------------------------------------------
extern "C" void kernel_launch(void* const* d_in, const int* in_sizes, int n_in,
                              void* d_out, int out_size, void* d_ws, size_t ws_size,
                              hipStream_t stream)
{
    const float* query = (const float*)d_in[0];
    const float* refp  = (const float*)d_in[1];
    const float* l2i   = (const float*)d_in[2];
    const float* Woff  = (const float*)d_in[3];
    const float* boff  = (const float*)d_in[4];
    const float* Wsw   = (const float*)d_in[5];
    const float* bsw   = (const float*)d_in[6];
    const float* f0    = (const float*)d_in[7];
    const float* f1    = (const float*)d_in[8];
    const float* f2    = (const float*)d_in[9];
    const float* f3    = (const float*)d_in[10];

    float* out = (float*)d_out;
    float* ws  = (float*)d_ws;

    float* off_out = ws;                               // Q*96   =   960000
    float* sw_out  = off_out + QN * NOFF;              // Q*128  =  1280000
    float* uvn     = sw_out  + QN * NSW;               // 320000*3
    float* wts     = uvn + QN * GN * PN * 3;           // 320000*4

    float* final_out = out;                            // 20,480,000 floats
    float* spc_out   = out + (long long)QN * GN * PN * CN;  // 640,000 floats

    const int mblocks = (QN / 16) / MTILES_PER_WAVE;   // 125
    dim3 g_off(mblocks, NOFF / 16);                    // 125 x 6
    dim3 g_sw (mblocks, NSW  / 16);                    // 125 x 8
    sbev_gemm_wmma<NOFF><<<g_off, 32, 0, stream>>>(query, Woff, boff, off_out);
    sbev_gemm_wmma<NSW ><<<g_sw,  32, 0, stream>>>(query, Wsw,  bsw,  sw_out);

    const int nproj = QN * GN * PN;                    // 320000
    sbev_proj<<<(nproj + 255) / 256, 256, 0, stream>>>(refp, l2i, off_out, sw_out,
                                                       uvn, wts, spc_out);

    const long long nsamp = (long long)QN * GN * PN * CN;   // 20,480,000
    sbev_sample<<<(unsigned)((nsamp + 255) / 256), 256, 0, stream>>>(
        f0, f1, f2, f3, uvn, wts, final_out);
}
